// Gemma3Attention_59940563583063
// MI455X (gfx1250) — matrix operand order repaired and checked by reading, not compiled, not run on hardware
//
#include <hip/hip_runtime.h>
#include <hip/hip_bf16.h>
#include <cstdint>
#include <cstddef>

// ---------------------------------------------------------------------------
// Gemma3 sliding-window attention for MI455X (gfx1250, wave32, WMMA bf16)
// GEMMs stage B panels through LDS with gfx1250 async global->LDS loads.
// ---------------------------------------------------------------------------

#define S_TOK 4096
#define HID   2304
#define NH    8
#define NKV   4
#define DH    256
#define GQ    2
#define WIN   1024
#define QSCALE 0.0625f   // 256^-0.5
#define TK    32         // K panel depth

typedef __attribute__((ext_vector_type(16))) __bf16 v16bf;
typedef __attribute__((ext_vector_type(8)))  __bf16 v8bf;
typedef __attribute__((ext_vector_type(8)))  float  v8f;

union AFrag { v16bf v; v8bf h[2]; };

__device__ __forceinline__ v8f wmma_bf16(const AFrag& a, const AFrag& b, v8f c) {
  return __builtin_amdgcn_wmma_f32_16x16x32_bf16(false, a.v, false, b.v,
                                                 (short)0, c, false, false);
}

// 32 contiguous bytes global -> LDS, async (ASYNCcnt), per-lane addresses.
// IOFFSET applies to both global and LDS addresses (ISA 08 §4.4).
__device__ __forceinline__ void async_copy32(unsigned lds_addr, const void* gptr) {
  unsigned long long ga = (unsigned long long)(uintptr_t)gptr;
  asm volatile("global_load_async_to_lds_b128 %0, %1, off\n\t"
               "global_load_async_to_lds_b128 %0, %1, off offset:16"
               :: "v"(lds_addr), "v"(ga) : "memory");
}

__device__ __forceinline__ void wait_async0() {
  asm volatile("s_wait_asynccnt 0" ::: "memory");
}

// ---------------------------------------------------------------------------
// Elementwise fp32 -> bf16 convert
// ---------------------------------------------------------------------------
__global__ void cvt_bf16_kernel(const float* __restrict__ in,
                                __bf16* __restrict__ out, size_t n) {
  size_t i = (size_t)blockIdx.x * blockDim.x + threadIdx.x;
  if (i < n) out[i] = (__bf16)in[i];
}

// W[K][N] fp32  ->  Wt[N][K] bf16  (so B fragments are contiguous along K)
__global__ void cvt_transpose_kernel(const float* __restrict__ W,
                                     __bf16* __restrict__ Wt, int K, int N) {
  size_t i = (size_t)blockIdx.x * blockDim.x + threadIdx.x;
  if (i >= (size_t)K * N) return;
  int n = (int)(i / K);
  int k = (int)(i % K);
  Wt[i] = (__bf16)W[(size_t)k * N + n];
}

// ---------------------------------------------------------------------------
// bf16 WMMA GEMM with async-LDS staged, double-buffered B panels:
//   C[M][N] (fp32) = A[M][K] * Bt[N][K]^T
// block = 128 threads (4 waves); block tile 64(M) x 64(N); K in panels of 32.
// B panel (64 rows x 32 k = 4KB) is async-staged once per block and shared by
// all 4 waves via LDS (removes the 4x duplicated global B traffic).
// Requires M%64==0, N%64==0, K%32==0.
// ---------------------------------------------------------------------------
__global__ __launch_bounds__(128)
void gemm_bf16_kernel(const __bf16* __restrict__ A,
                      const __bf16* __restrict__ Bt,
                      float* __restrict__ C, int M, int N, int K) {
  __shared__ __align__(16) __bf16 bsm[2][64 * TK];   // 2 x 4KB B panels

  const int tid  = threadIdx.x;
  const int lane = tid & 31;
  const int wv   = tid >> 5;
  const int m16  = lane & 15;
  const int kh   = lane >> 4;          // 16-lane half select
  const int m0   = blockIdx.y * 64 + wv * 16;
  const int n0   = blockIdx.x * 64;

  // Async staging assignment: 128 threads x 32B = 4KB panel.
  // thread t covers row = t>>1 (0..63), 32-byte half = t&1.
  const int srow  = tid >> 1;
  const int shalf = tid & 1;
  const __bf16* bsrc = Bt + (size_t)(n0 + srow) * K + shalf * 16; // +kb later
  const unsigned lds0 =
      (unsigned)(uintptr_t)(&bsm[0][0]) + (unsigned)(srow * (TK * 2) + shalf * 32);
  const unsigned bufstride = 64 * TK * 2;  // 4096 bytes

  const __bf16* arow = A + (size_t)(m0 + m16) * K;
  v8f acc0 = {}, acc1 = {}, acc2 = {}, acc3 = {};

  const int nk = K / TK;
  // preload panel 0
  async_copy32(lds0, bsrc);

  for (int i = 0; i < nk; ++i) {
    wait_async0();        // this wave's async stores to LDS are done
    __syncthreads();      // panel i visible to all 4 waves; buffer (i+1)&1 free
    if (i + 1 < nk)
      async_copy32(lds0 + ((unsigned)(i + 1) & 1u) * bufstride,
                   bsrc + (i + 1) * TK);

    const int kb = i * TK;
    AFrag a;
    // A 16x32 bf16 layout: lanes 0-15 hold K {0..15}, lanes 16-31 hold
    // K {16..31} for row m16.
    a.h[0] = *(const v8bf*)(arow + kb + kh * 16);
    a.h[1] = *(const v8bf*)(arow + kb + kh * 16 + 8);

    const __bf16* bp = &bsm[i & 1][0] + (size_t)m16 * TK + kh * 16;
    AFrag b;
#define GEMM_STEP(ACC, T)                                                   \
    b.h[0] = *(const v8bf*)(bp + (T) * 16 * TK);                            \
    b.h[1] = *(const v8bf*)(bp + (T) * 16 * TK + 8);                        \
    ACC = wmma_bf16(a, b, ACC);
    GEMM_STEP(acc0, 0)
    GEMM_STEP(acc1, 1)
    GEMM_STEP(acc2, 2)
    GEMM_STEP(acc3, 3)
#undef GEMM_STEP
  }

  // C layout: VGPR r -> row m0 + r + 8*kh, col n0 + t*16 + m16
  float* crow = C + (size_t)(m0 + kh * 8) * N + n0 + m16;
#pragma unroll
  for (int r = 0; r < 8; ++r) {
    crow[(size_t)r * N + 0]  = acc0[r];
    crow[(size_t)r * N + 16] = acc1[r];
    crow[(size_t)r * N + 32] = acc2[r];
    crow[(size_t)r * N + 48] = acc3[r];
  }
}

// ---------------------------------------------------------------------------
// RMSNorm + RoPE + scale + layout shuffles for q/k/v
// grid = S blocks of 256 threads (8 waves): wave w does q head w; waves 0-3
// also do k head w; waves 4-7 transpose v head (w-4).
// ---------------------------------------------------------------------------
__device__ __forceinline__ float wave_sum32(float v) {
#pragma unroll
  for (int m = 16; m >= 1; m >>= 1) v += __shfl_xor(v, m, 32);
  return v;
}

__device__ __forceinline__ void norm_rope_store(const float* __restrict__ src,
                                                const float* __restrict__ nw,
                                                float pos, float outScale,
                                                __bf16* __restrict__ dst,
                                                int lane) {
  float x[8];
  const int d0 = lane * 8;
#pragma unroll
  for (int j = 0; j < 8; ++j) x[j] = src[d0 + j];
  float ss = 0.f;
#pragma unroll
  for (int j = 0; j < 8; ++j) ss += x[j] * x[j];
  ss = wave_sum32(ss);
  const float rstd = rsqrtf(ss * (1.0f / DH) + 1e-6f);
#pragma unroll
  for (int j = 0; j < 8; ++j) x[j] *= rstd * (1.0f + nw[d0 + j]);
  // RoPE: halves are lanes 0-15 (d<128) and lanes 16-31 (d>=128); swap via
  // shfl_xor(16). inv_freq[i] = 10000^(-i/128).
#pragma unroll
  for (int j = 0; j < 8; ++j) {
    const int fi = (d0 + j) & 127;
    const float fr = pos * powf(10000.0f, -(float)fi * (1.0f / 128.0f));
    const float cs = cosf(fr);
    const float sn = sinf(fr);
    const float other = __shfl_xor(x[j], 16, 32);
    const float o = (lane < 16) ? (x[j] * cs - other * sn)
                                : (other * sn + x[j] * cs);
    dst[d0 + j] = (__bf16)(o * outScale);
  }
}

__global__ __launch_bounds__(256)
void qkv_post_kernel(const float* __restrict__ qf, const float* __restrict__ kf,
                     const float* __restrict__ vf, const float* __restrict__ qnw,
                     const float* __restrict__ knw, const int* __restrict__ pos_ids,
                     __bf16* __restrict__ qb,   // [S][NH][DH]
                     __bf16* __restrict__ kb,   // [NKV][S][DH]
                     __bf16* __restrict__ vt) { // [NKV][DH][S]
  const int s    = blockIdx.x;
  const int wv   = threadIdx.x >> 5;
  const int lane = threadIdx.x & 31;
  const float pos = (float)pos_ids[s];

  // q: head h = wv
  norm_rope_store(qf + ((size_t)s * NH + wv) * DH, qnw, pos, QSCALE,
                  qb + ((size_t)s * NH + wv) * DH, lane);

  if (wv < NKV) {
    const int c = wv;
    norm_rope_store(kf + ((size_t)s * NKV + c) * DH, knw, pos, 1.0f,
                    kb + ((size_t)c * S_TOK + s) * DH, lane);
  } else {
    const int c = wv - NKV;
    const float* src = vf + ((size_t)s * NKV + c) * DH;
    const int d0 = lane * 8;
#pragma unroll
    for (int j = 0; j < 8; ++j)
      vt[((size_t)c * DH + d0 + j) * S_TOK + s] = (__bf16)src[d0 + j];
  }
}

// ---------------------------------------------------------------------------
// Flash attention with 1024 sliding window. One wave per (head, 16-query
// tile). Scores via 16 WMMA / 32 keys; probs staged through LDS to A-layout;
// P*V via 16 WMMA / 32 keys. fp32 online-softmax state in C-layout.
// ---------------------------------------------------------------------------
__global__ __launch_bounds__(32)
void attn_kernel(const __bf16* __restrict__ qb, const __bf16* __restrict__ kb,
                 const __bf16* __restrict__ vt, __bf16* __restrict__ ob) {
  __shared__ __align__(16) __bf16 pbuf[16 * 32];
  const int lane = threadIdx.x & 31;
  const int m16  = lane & 15;
  const int kh   = lane >> 4;
  const int h    = blockIdx.y;
  const int c    = h / GQ;
  const int q0   = blockIdx.x * 16;

  // Q tile A-fragments for full D=256 (8 fragments of 16x32)
  AFrag qa[8];
  const __bf16* qrow = qb + ((size_t)(q0 + m16) * NH + h) * DH;
#pragma unroll
  for (int f = 0; f < 8; ++f) {
    qa[f].h[0] = *(const v8bf*)(qrow + f * 32 + kh * 16);
    qa[f].h[1] = *(const v8bf*)(qrow + f * 32 + kh * 16 + 8);
  }

  v8f acc[16];
#pragma unroll
  for (int t = 0; t < 16; ++t) acc[t] = (v8f){};
  float mrow[8], lrow[8];
#pragma unroll
  for (int r = 0; r < 8; ++r) { mrow[r] = -1e30f; lrow[r] = 0.f; }

  const int tlo  = (q0 >= (WIN - 1)) ? (q0 - (WIN - 1)) : 0;
  const int t0   = tlo & ~31;
  const int tend = q0 + 16;

  for (int tb = t0; tb < tend; tb += 32) {
    // ---- scores: two 16x16 tiles over keys [tb, tb+31] ----
    v8f s0 = {}, s1 = {};
    const __bf16* kr0 = kb + ((size_t)c * S_TOK + tb + m16) * DH + kh * 16;
    const __bf16* kr1 = kr0 + (size_t)16 * DH;
#pragma unroll
    for (int f = 0; f < 8; ++f) {
      AFrag b0, b1;
      b0.h[0] = *(const v8bf*)(kr0 + f * 32);
      b0.h[1] = *(const v8bf*)(kr0 + f * 32 + 8);
      b1.h[0] = *(const v8bf*)(kr1 + f * 32);
      b1.h[1] = *(const v8bf*)(kr1 + f * 32 + 8);
      s0 = wmma_bf16(qa[f], b0, s0);
      s1 = wmma_bf16(qa[f], b1, s1);
    }

    // ---- online softmax in C-layout ----
    const int tc0 = tb + m16;
    const int tc1 = tb + 16 + m16;
#pragma unroll
    for (int r = 0; r < 8; ++r) {
      const int qi = q0 + r + 8 * kh;
      float v0 = ((tc0 <= qi) && (qi - tc0 < WIN)) ? s0[r] : -1e30f;
      float v1 = ((tc1 <= qi) && (qi - tc1 < WIN)) ? s1[r] : -1e30f;
      float mx = fmaxf(v0, v1);
#pragma unroll
      for (int o = 1; o < 16; o <<= 1) mx = fmaxf(mx, __shfl_xor(mx, o, 32));
      const float mnew = fmaxf(mrow[r], mx);
      const float resc = __expf(mrow[r] - mnew);
      const float p0 = __expf(v0 - mnew);
      const float p1 = __expf(v1 - mnew);
      float ps = p0 + p1;
#pragma unroll
      for (int o = 1; o < 16; o <<= 1) ps += __shfl_xor(ps, o, 32);
      lrow[r] = lrow[r] * resc + ps;
      mrow[r] = mnew;
#pragma unroll
      for (int t = 0; t < 16; ++t) acc[t][r] *= resc;
      pbuf[(r + 8 * kh) * 32 + m16]      = (__bf16)p0;
      pbuf[(r + 8 * kh) * 32 + 16 + m16] = (__bf16)p1;
    }
    __syncthreads();

    // ---- reload probs in A-layout, P(16x32) * V(32x256) ----
    AFrag pa;
    pa.h[0] = *(const v8bf*)(pbuf + m16 * 32 + kh * 16);
    pa.h[1] = *(const v8bf*)(pbuf + m16 * 32 + kh * 16 + 8);
    const __bf16* vr = vt + ((size_t)c * DH + m16) * S_TOK + tb + kh * 16;
#pragma unroll
    for (int t = 0; t < 16; ++t) {
      AFrag b;
      b.h[0] = *(const v8bf*)(vr + (size_t)t * 16 * S_TOK);
      b.h[1] = *(const v8bf*)(vr + (size_t)t * 16 * S_TOK + 8);
      acc[t] = wmma_bf16(pa, b, acc[t]);
    }
    __syncthreads();
  }

  // ---- epilogue: divide by l, store bf16 [S][NH*DH] ----
#pragma unroll
  for (int r = 0; r < 8; ++r) {
    const float inv = 1.0f / lrow[r];
    const size_t row = ((size_t)(q0 + r + 8 * kh) * NH + h) * DH;
#pragma unroll
    for (int t = 0; t < 16; ++t)
      ob[row + t * 16 + m16] = (__bf16)(acc[t][r] * inv);
  }
}

// ---------------------------------------------------------------------------
// Host-side launcher
// ---------------------------------------------------------------------------
extern "C" void kernel_launch(void* const* d_in, const int* in_sizes, int n_in,
                              void* d_out, int out_size, void* d_ws, size_t ws_size,
                              hipStream_t stream) {
  const float* x   = (const float*)d_in[0];
  const int*   pos = (const int*)d_in[1];
  const float* wq  = (const float*)d_in[2];
  const float* wk  = (const float*)d_in[3];
  const float* wv  = (const float*)d_in[4];
  const float* wo  = (const float*)d_in[5];
  const float* qnw = (const float*)d_in[6];
  const float* knw = (const float*)d_in[7];
  float* out = (float*)d_out;

  char* base = (char*)d_ws;
  size_t off = 0;
  auto alloc = [&](size_t bytes) -> void* {
    void* r = base + off;
    off = (off + bytes + 255) & ~(size_t)255;
    return r;
  };

  __bf16* xb  = (__bf16*)alloc((size_t)S_TOK * HID * 2);
  __bf16* wqT = (__bf16*)alloc((size_t)NH  * DH * HID * 2);
  __bf16* wkT = (__bf16*)alloc((size_t)NKV * DH * HID * 2);
  __bf16* wvT = (__bf16*)alloc((size_t)NKV * DH * HID * 2);
  __bf16* woT = (__bf16*)alloc((size_t)HID * NH * DH * 2);
  float*  qf  = (float*) alloc((size_t)S_TOK * NH  * DH * 4);
  float*  kf  = (float*) alloc((size_t)S_TOK * NKV * DH * 4);
  float*  vf  = (float*) alloc((size_t)S_TOK * NKV * DH * 4);
  __bf16* qb  = (__bf16*)alloc((size_t)S_TOK * NH  * DH * 2);
  __bf16* kb  = (__bf16*)alloc((size_t)NKV * S_TOK * DH * 2);
  __bf16* vt  = (__bf16*)alloc((size_t)NKV * DH * S_TOK * 2);
  __bf16* ab  = (__bf16*)alloc((size_t)S_TOK * NH  * DH * 2);

  // 1) convert inputs to bf16 (weights transposed to [N][K])
  {
    size_t n = (size_t)S_TOK * HID;
    cvt_bf16_kernel<<<(unsigned)((n + 255) / 256), 256, 0, stream>>>(x, xb, n);
  }
  {
    size_t n = (size_t)HID * NH * DH;
    cvt_transpose_kernel<<<(unsigned)((n + 255) / 256), 256, 0, stream>>>(wq, wqT, HID, NH * DH);
  }
  {
    size_t n = (size_t)HID * NKV * DH;
    cvt_transpose_kernel<<<(unsigned)((n + 255) / 256), 256, 0, stream>>>(wk, wkT, HID, NKV * DH);
    cvt_transpose_kernel<<<(unsigned)((n + 255) / 256), 256, 0, stream>>>(wv, wvT, HID, NKV * DH);
  }
  {
    size_t n = (size_t)NH * DH * HID;
    cvt_transpose_kernel<<<(unsigned)((n + 255) / 256), 256, 0, stream>>>(wo, woT, NH * DH, HID);
  }

  // 2) QKV projections (WMMA bf16 -> fp32, async-LDS staged B)
  {
    dim3 g(NH * DH / 64, S_TOK / 64);
    gemm_bf16_kernel<<<g, 128, 0, stream>>>(xb, wqT, qf, S_TOK, NH * DH, HID);
  }
  {
    dim3 g(NKV * DH / 64, S_TOK / 64);
    gemm_bf16_kernel<<<g, 128, 0, stream>>>(xb, wkT, kf, S_TOK, NKV * DH, HID);
    gemm_bf16_kernel<<<g, 128, 0, stream>>>(xb, wvT, vf, S_TOK, NKV * DH, HID);
  }

  // 3) RMSNorm + RoPE + scale + re-layout
  qkv_post_kernel<<<S_TOK, 256, 0, stream>>>(qf, kf, vf, qnw, knw, pos, qb, kb, vt);

  // 4) sliding-window flash attention
  {
    dim3 g(S_TOK / 16, NH);
    attn_kernel<<<g, 32, 0, stream>>>(qb, kb, vt, ab);
  }

  // 5) output projection -> fp32 d_out
  {
    dim3 g(HID / 64, S_TOK / 64);
    gemm_bf16_kernel<<<g, 128, 0, stream>>>(ab, woT, out, S_TOK, HID, NH * DH);
  }
}
